// SlidingWindowAttention_1219770712403
// MI455X (gfx1250) — matrix operand order repaired and checked by reading, not compiled, not run on hardware
//
#include <hip/hip_runtime.h>

typedef __attribute__((ext_vector_type(16))) _Float16 v16h;
typedef __attribute__((ext_vector_type(8)))  float    v8f;

#define SEQ      2048
#define DHEAD    128
#define BLOCK_M  128
#define KTILE    32
#define KLDS_STRIDE 132   // halfs per K-tile row (128 + pad)
#define VT_STRIDE    36   // halfs per Vt row (32 + pad)
#define P_STRIDE     36   // halfs per P row (32 + pad)

// Barrier that makes LDS staging visible (dscnt==0) but leaves outstanding
// global prefetch loads in flight across the barrier (unlike __syncthreads(),
// which lowers to s_wait_loadcnt_dscnt 0x0 and drains the prefetch early).
#define LDS_BARRIER() asm volatile(            \
    "s_wait_dscnt 0\n\t"                       \
    "s_barrier_signal -1\n\t"                  \
    "s_barrier_wait -1" ::: "memory")

__global__ __launch_bounds__(256, 1)
void fa_fwd_causal(const float* __restrict__ Q,
                   const float* __restrict__ K,
                   const float* __restrict__ V,
                   float* __restrict__ O)
{
    __shared__ _Float16 Klds[2][KTILE * KLDS_STRIDE];   // K tile, row-major [key][d], double-buffered
    __shared__ _Float16 Vtlds[2][DHEAD * VT_STRIDE];    // V tile, transposed [d][key], double-buffered
    __shared__ _Float16 Plds[8][16 * P_STRIDE];         // per-wave P staging [row][key]

    const int tid  = threadIdx.x;
    const int lane = tid & 31;
    const int wave = tid >> 5;
    const int half = lane >> 4;   // which 16-lane half
    const int l16  = lane & 15;

    const int qblock = blockIdx.x * BLOCK_M;
    const int bh     = blockIdx.y;
    const size_t base = (size_t)bh * SEQ * DHEAD;

    const int wq = qblock + wave * 16;            // first query row of this wave

    const float scale = 0.08838834764831845f;     // 1/sqrt(128)

    // ---- Q fragments in WMMA A layout (16x32 per k-step), fp32 -> f16, scale folded in
    v16h aq[4];
    {
        const float* qrow = Q + base + (size_t)(wq + l16) * DHEAD;
        #pragma unroll
        for (int t = 0; t < 4; ++t) {
            #pragma unroll
            for (int j = 0; j < 16; ++j) {
                const int d = 32 * t + j + (half << 4);
                aq[t][j] = (_Float16)(qrow[d] * scale);
            }
        }
    }

    v8f oacc[8];
    #pragma unroll
    for (int c = 0; c < 8; ++c) oacc[c] = (v8f){};
    float mrow[8], lrow[8];
    #pragma unroll
    for (int r = 0; r < 8; ++r) { mrow[r] = -INFINITY; lrow[r] = 0.0f; }

    const int ntiles = (qblock + BLOCK_M) / KTILE;   // causal bound for the whole block (>= 4)

    // ---- staging assignment: 256 threads cover a 32x128 fp32 tile
    const int key = tid >> 3;            // 0..31
    const int d0  = (tid & 7) << 4;      // 0,16,...,112

    // ---- prologue: tile 0 in flight in registers
    float kreg[16], vreg[16];
    {
        const float* kp = K + base + (size_t)key * DHEAD + d0;
        const float* vp = V + base + (size_t)key * DHEAD + d0;
        #pragma unroll
        for (int i = 0; i < 16; ++i) { kreg[i] = kp[i]; vreg[i] = vp[i]; }
    }

    for (int kt = 0; kt < ntiles; ++kt) {
        const int ko = kt * KTILE;
        const int b  = kt & 1;

        // ---- drain held tile into LDS buffer b (waits only on its own global loads)
        {
            _Float16* kdst = &Klds[b][key * KLDS_STRIDE + d0];
            #pragma unroll
            for (int i = 0; i < 16; ++i) kdst[i] = (_Float16)kreg[i];
            #pragma unroll
            for (int i = 0; i < 16; ++i)
                Vtlds[b][(d0 + i) * VT_STRIDE + key] = (_Float16)vreg[i];
        }

        // ---- issue next tile's global loads; they stay in flight through the
        //      barrier and this tile's compute (consumed at the next ds_store)
        if (kt + 1 < ntiles) {
            const float* kp = K + base + (size_t)(ko + KTILE + key) * DHEAD + d0;
            const float* vp = V + base + (size_t)(ko + KTILE + key) * DHEAD + d0;
            #pragma unroll
            for (int i = 0; i < 16; ++i) { kreg[i] = kp[i]; vreg[i] = vp[i]; }
        }

        LDS_BARRIER();   // staging visible to all waves; prefetch loads NOT drained

        if (ko > wq + 15) continue;   // wave-uniform skip: tile entirely above diagonal

        // ---- GEMM1: scores[16x32] = Q_tile x K_tile^T  (two 16-key subtiles)
        v8f sc[2];
        #pragma unroll
        for (int n = 0; n < 2; ++n) {
            v8f acc = (v8f){};
            #pragma unroll
            for (int t = 0; t < 4; ++t) {
                v16h bk;   // B layout: element j -> K(d) = j + 16*half (+32t), N(key) = l16
                const _Float16* kr = &Klds[b][(16 * n + l16) * KLDS_STRIDE + 32 * t + (half << 4)];
                #pragma unroll
                for (int j = 0; j < 16; ++j) bk[j] = kr[j];
                acc = __builtin_amdgcn_wmma_f32_16x16x32_f16(false, aq[t], false, bk,
                                                             (short)0, acc, false, false);
            }
            sc[n] = acc;
        }

        // ---- online softmax (rows live per 16-lane half in D layout)
        #pragma unroll
        for (int r = 0; r < 8; ++r) {
            const int qrow = wq + r + (half << 3);
            float s0 = (ko + l16      <= qrow) ? sc[0][r] : -INFINITY;
            float s1 = (ko + 16 + l16 <= qrow) ? sc[1][r] : -INFINITY;
            float mx = fmaxf(s0, s1);
            #pragma unroll
            for (int m = 1; m < 16; m <<= 1)
                mx = fmaxf(mx, __shfl_xor(mx, m, 32));
            const float mnew = fmaxf(mrow[r], mx);
            const float p0 = __expf(s0 - mnew);
            const float p1 = __expf(s1 - mnew);
            float rs = p0 + p1;
            #pragma unroll
            for (int m = 1; m < 16; m <<= 1)
                rs += __shfl_xor(rs, m, 32);
            const float corr = __expf(mrow[r] - mnew);
            lrow[r] = lrow[r] * corr + rs;
            mrow[r] = mnew;
            #pragma unroll
            for (int c = 0; c < 8; ++c) oacc[c][r] *= corr;
            // stash P in row-major layout for A-fragment reload
            Plds[wave][(r + (half << 3)) * P_STRIDE + l16]      = (_Float16)p0;
            Plds[wave][(r + (half << 3)) * P_STRIDE + 16 + l16] = (_Float16)p1;
        }

        // same-wave DS store->load ordering
        asm volatile("s_wait_dscnt 0" ::: "memory");

        // ---- reload P as WMMA A fragment (16 rows x 32 keys)
        v16h ap;
        {
            const _Float16* pr = &Plds[wave][l16 * P_STRIDE];
            #pragma unroll
            for (int j = 0; j < 16; ++j) {
                const int kk = j + (half << 4);
                ap[j] = pr[kk];
            }
        }

        // ---- GEMM2: O[16x128] += P[16x32] x V[32x128]
        #pragma unroll
        for (int c = 0; c < 8; ++c) {
            v16h bv;   // B layout from transposed V: element j -> K(key) = j + 16*half
            const _Float16* vr = &Vtlds[b][(16 * c + l16) * VT_STRIDE + (half << 4)];
            #pragma unroll
            for (int j = 0; j < 16; ++j) bv[j] = vr[j];
            oacc[c] = __builtin_amdgcn_wmma_f32_16x16x32_f16(false, ap, false, bv,
                                                             (short)0, oacc[c], false, false);
        }
    }

    // ---- epilogue: O /= l, fp32 store (16 consecutive floats per half-wave per store)
    #pragma unroll
    for (int r = 0; r < 8; ++r) {
        const float inv  = 1.0f / lrow[r];
        const int   qrow = wq + r + (half << 3);
        float* op = O + base + (size_t)qrow * DHEAD + l16;
        #pragma unroll
        for (int c = 0; c < 8; ++c)
            op[16 * c] = oacc[c][r] * inv;
    }
}

extern "C" void kernel_launch(void* const* d_in, const int* in_sizes, int n_in,
                              void* d_out, int out_size, void* d_ws, size_t ws_size,
                              hipStream_t stream) {
    const float* Q = (const float*)d_in[0];
    const float* K = (const float*)d_in[1];
    const float* V = (const float*)d_in[2];
    float* O = (float*)d_out;
    const int BH = in_sizes[0] / (SEQ * DHEAD);   // = 4*16 = 64
    dim3 grid(SEQ / BLOCK_M, BH);
    fa_fwd_causal<<<grid, 256, 0, stream>>>(Q, K, V, O);
}
